// GatedDeltaNetAggregator_79302276153501
// MI455X (gfx1250) — matrix operand order repaired and checked by reading, not compiled, not run on hardware
//
#include <hip/hip_runtime.h>
#include <hip/hip_bf16.h>
#include <math.h>

typedef _Float16 half_t;
typedef __attribute__((ext_vector_type(16))) _Float16 v16h;
typedef __attribute__((ext_vector_type(8)))  float    v8f;

#define B_SZ    4
#define T_SZ    4096
#define D_SZ    1024
#define H_SZ    16
#define DK_SZ   64
#define DV_SZ   128
#define KEY_DIM 1024
#define VAL_DIM 2048
#define NTOK    (B_SZ * T_SZ)      // 16384
#define NPAD    4224               // 33 * 128 (padded fused-N for tiling)
#define CONVW   4

// CDNA5 async global->LDS path (ASYNCcnt), guarded so compile never regresses.
#if defined(__AMDGCN__) && __has_builtin(__builtin_amdgcn_global_load_async_to_lds_b128)
#define USE_ASYNC_LDS 1
#else
#define USE_ASYNC_LDS 0
#endif

#if USE_ASYNC_LDS
typedef int v4i_vec __attribute__((vector_size(16)));
typedef __attribute__((address_space(1))) v4i_vec* g128_t;
typedef __attribute__((address_space(3))) v4i_vec* l128_t;

__device__ __forceinline__ void async_copy16(const half_t* g, half_t* l) {
  __builtin_amdgcn_global_load_async_to_lds_b128(
      (g128_t)const_cast<half_t*>(g), (l128_t)l, 0, 0);
}
__device__ __forceinline__ void async_wait0() {
#if __has_builtin(__builtin_amdgcn_s_wait_asynccnt)
  __builtin_amdgcn_s_wait_asynccnt(0);
#else
  asm volatile("s_wait_asynccnt 0" ::: "memory");
#endif
}
#endif

// ---------------------------------------------------------------------------
// k0: convert z to f16, build fused (zero-padded) f16 weight [D][NPAD]
// cols: [Wq 0..1023 | Wk 1024..2047 | Wv 2048..4095 | Wb 4096..4111 | Wa 4112..4127 | pad]
// ---------------------------------------------------------------------------
__global__ void k0_convert(const float* __restrict__ z,
                           const float* __restrict__ Wq, const float* __restrict__ Wk,
                           const float* __restrict__ Wv, const float* __restrict__ Wb,
                           const float* __restrict__ Wa,
                           half_t* __restrict__ zf, half_t* __restrict__ wf) {
  const int stride = gridDim.x * blockDim.x;
  int idx = blockIdx.x * blockDim.x + threadIdx.x;
  const int totZ = NTOK * D_SZ;
  for (int i = idx; i < totZ; i += stride) zf[i] = (half_t)z[i];
  const int totW = D_SZ * NPAD;
  for (int i = idx; i < totW; i += stride) {
    int d = i / NPAD, n = i - d * NPAD;
    float v = 0.f;
    if (n < 1024)       v = Wq[(size_t)d * KEY_DIM + n];
    else if (n < 2048)  v = Wk[(size_t)d * KEY_DIM + (n - 1024)];
    else if (n < 4096)  v = Wv[(size_t)d * VAL_DIM + (n - 2048)];
    else if (n < 4112)  v = Wb[(size_t)d * H_SZ + (n - 4096)];
    else if (n < 4128)  v = Wa[(size_t)d * H_SZ + (n - 4112)];
    wf[i] = (half_t)v;
  }
}

// ---------------------------------------------------------------------------
// k1: WMMA GEMM  C[NTOK][NPAD](f16) = A[NTOK][D](f16) x W[D][NPAD](f16)
// block: 256 thr = 8 waves; tile 128x128; K-tile 64 (16 WMMA per stage/barrier)
// A tile staged via async global->LDS (ASYNCcnt path) when available.
// ---------------------------------------------------------------------------
#define KT  64
#define LDA 72   // KT + 8 pad (halfs); 144B row stride, 16B aligned
#define LDB 72

__device__ __forceinline__ v16h frag_a_from_lds(const half_t* base) {
  // A 16x32 f16 fragment per ISA 7.12.2: lanes 0-15 -> K {0..7,16..23},
  // lanes 16-31 -> K {8..15,24..31}; K-consecutive pairs per VGPR.
  const int lane = threadIdx.x & 31;
  const half_t* rp = base + (lane & 15) * LDA + ((lane >> 4) << 3);
  v16h f;
  uint32_t* fp = reinterpret_cast<uint32_t*>(&f);
#pragma unroll
  for (int p = 0; p < 8; ++p) {
    const int kb = (p < 4) ? (p << 1) : (16 + ((p - 4) << 1));
    fp[p] = *reinterpret_cast<const uint32_t*>(rp + kb);
  }
  return f;
}

__device__ __forceinline__ v16h frag_b_from_ldsT(const half_t* base) {
  // B 32x16 f16 fragment from n-major (transposed) LDS tile:
  // lane -> N = lane&15; lanes 0-15 K {0..7,16..23}, lanes 16-31 K {8..15,24..31}.
  const int lane = threadIdx.x & 31;
  const half_t* rp = base + (lane & 15) * LDB + ((lane >> 4) << 3);
  v16h f;
  uint32_t* fp = reinterpret_cast<uint32_t*>(&f);
#pragma unroll
  for (int p = 0; p < 8; ++p)
    fp[p] = *reinterpret_cast<const uint32_t*>(rp + ((p < 4) ? (p << 1) : (16 + ((p - 4) << 1))));
  return f;
}

__global__ __launch_bounds__(256)
void k1_gemm_fused(const half_t* __restrict__ A, const half_t* __restrict__ W,
                   half_t* __restrict__ C) {
  __shared__ half_t As[128 * LDA];
  __shared__ half_t Bs[128 * LDB];
  const int tid = threadIdx.x;
  const int wid = tid >> 5;
  const int wm  = wid >> 2;         // 0..1
  const int wn  = wid & 3;          // 0..3
  const int bm  = blockIdx.y * 128;
  const int bn  = blockIdx.x * 128;

  v8f acc[4][2] = {};

  for (int kt = 0; kt < D_SZ; kt += KT) {
    // ---- stage A tile 128xKT (row-major, padded stride) ----
#if USE_ASYNC_LDS
#pragma unroll
    for (int c = 0; c < 4; ++c) {
      const int chunk = tid * 4 + c;           // 0..1023
      const int row   = chunk >> 3;            // 0..127
      const int kc    = (chunk & 7) << 3;      // 0..56
      async_copy16(A + (size_t)(bm + row) * D_SZ + kt + kc,
                   As + row * LDA + kc);
    }
#else
#pragma unroll
    for (int c = 0; c < 4; ++c) {
      const int chunk = tid * 4 + c;
      const int row   = chunk >> 3;
      const int kc    = (chunk & 7) << 3;
      const uint4 dv = *reinterpret_cast<const uint4*>(
          A + (size_t)(bm + row) * D_SZ + kt + kc);
      uint32_t* dst = reinterpret_cast<uint32_t*>(As + row * LDA + kc);
      dst[0] = dv.x; dst[1] = dv.y; dst[2] = dv.z; dst[3] = dv.w;
    }
#endif
    // ---- stage B tile KTx128, transposed into [n][k] ----
#pragma unroll
    for (int c = 0; c < 4; ++c) {
      const int chunk = tid * 4 + c;           // 0..1023
      const int kr    = chunk >> 4;            // 0..63
      const int nc    = (chunk & 15) << 3;     // 0..120
      const uint4 dv = *reinterpret_cast<const uint4*>(
          W + (size_t)(kt + kr) * NPAD + bn + nc);
      const half_t* hv = reinterpret_cast<const half_t*>(&dv);
#pragma unroll
      for (int j = 0; j < 8; ++j) Bs[(nc + j) * LDB + kr] = hv[j];
    }
#if USE_ASYNC_LDS
    async_wait0();
#endif
    __syncthreads();

    // ---- compute: 2 K-steps x (4M x 2N) WMMA ----
#pragma unroll
    for (int ks = 0; ks < KT; ks += 32) {
      v16h af[4];
#pragma unroll
      for (int mi = 0; mi < 4; ++mi)
        af[mi] = frag_a_from_lds(As + (wm * 64 + mi * 16) * LDA + ks);
#pragma unroll
      for (int ni = 0; ni < 2; ++ni) {
        const v16h bf = frag_b_from_ldsT(Bs + (wn * 32 + ni * 16) * LDB + ks);
#pragma unroll
        for (int mi = 0; mi < 4; ++mi)
          acc[mi][ni] = __builtin_amdgcn_wmma_f32_16x16x32_f16(
              false, af[mi], false, bf, (short)0, acc[mi][ni], false, false);
      }
    }
    __syncthreads();
  }

  // epilogue: C/D layout -> N = lane&15, M = r + 8*(lane>=16)
  const int lane = tid & 31;
  const int nloc = lane & 15;
  const int mhi  = (lane >> 4) << 3;
#pragma unroll
  for (int mi = 0; mi < 4; ++mi)
#pragma unroll
    for (int ni = 0; ni < 2; ++ni)
#pragma unroll
      for (int r = 0; r < 8; ++r) {
        const int row = bm + wm * 64 + mi * 16 + mhi + r;
        const int col = bn + wn * 32 + ni * 16 + nloc;
        C[(size_t)row * NPAD + col] = (half_t)acc[mi][ni][r];
      }
}

// ---------------------------------------------------------------------------
// k2: per-token conv(width 4)+SiLU, per-head L2 norm for q/k, v, beta, g
// outputs head-major: q/k [B*H][T][DK] f16, v [B*H][T][DV] f16, g/beta [B*H][T] f32
// ---------------------------------------------------------------------------
__global__ __launch_bounds__(256)
void k2_pointwise(const half_t* __restrict__ pre,
                  const float* __restrict__ cq, const float* __restrict__ ck,
                  const float* __restrict__ cv,
                  const float* __restrict__ A_log, const float* __restrict__ dt_bias,
                  half_t* __restrict__ qo, half_t* __restrict__ ko,
                  half_t* __restrict__ vo,
                  float* __restrict__ go, float* __restrict__ bo) {
  __shared__ float sbuf[4096];
  __shared__ float rsq[32];
  const int m = blockIdx.x;            // token
  const int b = m / T_SZ;
  const int t = m - b * T_SZ;
  const int tid = threadIdx.x;

  for (int c = tid; c < 4096; c += 256) {
    const float* w;
    if (c < 1024)      w = cq + (size_t)c * CONVW;
    else if (c < 2048) w = ck + (size_t)(c - 1024) * CONVW;
    else               w = cv + (size_t)(c - 2048) * CONVW;
    float y = 0.f;
#pragma unroll
    for (int i = 0; i < CONVW; ++i) {
      const int ts = t - (CONVW - 1) + i;
      if (ts >= 0)
        y = fmaf(w[i], (float)pre[(size_t)(b * T_SZ + ts) * NPAD + c], y);
    }
    sbuf[c] = y * (1.f / (1.f + __expf(-y)));   // silu
  }
  __syncthreads();
  if (tid < 32) {
    const int base = (tid < 16) ? tid * 64 : 1024 + (tid - 16) * 64;
    float s = 0.f;
    for (int i = 0; i < 64; ++i) { const float x = sbuf[base + i]; s += x * x; }
    rsq[tid] = rsqrtf(s + 1e-6f);
  }
  __syncthreads();

  const float qscale = 0.125f;  // DK^-0.5
  for (int c = tid; c < 1024; c += 256) {
    const int h = c >> 6, dk = c & 63;
    const size_t o = ((size_t)(b * H_SZ + h) * T_SZ + t) * DK_SZ + dk;
    qo[o] = (half_t)(sbuf[c] * rsq[h] * qscale);
    ko[o] = (half_t)(sbuf[1024 + c] * rsq[16 + h]);
  }
  for (int c = tid; c < 2048; c += 256) {
    const int h = c >> 7, dv = c & 127;
    vo[((size_t)(b * H_SZ + h) * T_SZ + t) * DV_SZ + dv] = (half_t)sbuf[2048 + c];
  }
  if (tid < H_SZ) {
    const float pb = (float)pre[(size_t)m * NPAD + 4096 + tid];
    bo[(size_t)(b * H_SZ + tid) * T_SZ + t] = 1.f / (1.f + __expf(-pb));
    const float pa = (float)pre[(size_t)m * NPAD + 4112 + tid] + dt_bias[tid];
    const float sp = (pa > 20.f) ? pa : log1pf(__expf(pa));
    go[(size_t)(b * H_SZ + tid) * T_SZ + t] = -__expf(A_log[tid]) * sp;
  }
}

// ---------------------------------------------------------------------------
// k3: gated delta-rule scan; state column slice lives in VGPRs (64 f32/thread)
// grid = B*H*NSPLIT blocks of wave32; each block owns 32 DV columns
// ---------------------------------------------------------------------------
#define NSPLIT 4
#define CHNK   16
__global__ __launch_bounds__(32)
void k3_scan(const half_t* __restrict__ q, const half_t* __restrict__ k,
             const half_t* __restrict__ v, const float* __restrict__ g,
             const float* __restrict__ beta, float* __restrict__ olast) {
  __shared__ float sk[CHNK][DK_SZ];
  __shared__ float sq[CHNK][DK_SZ];
  __shared__ float sg[CHNK], sb[CHNK];
  const int blk  = blockIdx.x;
  const int bh   = blk / NSPLIT;
  const int sp   = blk - bh * NSPLIT;
  const int lane = threadIdx.x;
  const int col  = sp * 32 + lane;
  const half_t* kbase = k + (size_t)bh * T_SZ * DK_SZ;
  const half_t* qbase = q + (size_t)bh * T_SZ * DK_SZ;
  const half_t* vbase = v + (size_t)bh * T_SZ * DV_SZ;
  const float*  gbase = g + (size_t)bh * T_SZ;
  const float*  bbase = beta + (size_t)bh * T_SZ;

  float S[DK_SZ];
#pragma unroll
  for (int i = 0; i < DK_SZ; ++i) S[i] = 0.f;

  for (int t0 = 0; t0 < T_SZ; t0 += CHNK) {
    __syncthreads();
    for (int idx = lane; idx < CHNK * DK_SZ; idx += 32) {
      const int tt = idx >> 6, dk = idx & 63;
      sk[tt][dk] = (float)kbase[(size_t)(t0 + tt) * DK_SZ + dk];
      sq[tt][dk] = (float)qbase[(size_t)(t0 + tt) * DK_SZ + dk];
    }
    if (lane < CHNK) { sg[lane] = gbase[t0 + lane]; sb[lane] = bbase[t0 + lane]; }
    __syncthreads();
    if (t0 + CHNK < T_SZ) {
      __builtin_prefetch(kbase + (size_t)(t0 + CHNK) * DK_SZ, 0, 1);
      __builtin_prefetch(qbase + (size_t)(t0 + CHNK) * DK_SZ, 0, 1);
    }
    for (int tt = 0; tt < CHNK; ++tt) {
      const int   t  = t0 + tt;
      const float eg = __expf(sg[tt]);
      const float vj = (float)vbase[(size_t)t * DV_SZ + col];
      float mem = 0.f;
#pragma unroll
      for (int i = 0; i < DK_SZ; ++i) { S[i] *= eg; mem = fmaf(sk[tt][i], S[i], mem); }
      const float upd = sb[tt] * (vj - mem);
#pragma unroll
      for (int i = 0; i < DK_SZ; ++i) S[i] = fmaf(sk[tt][i], upd, S[i]);
      if (t == T_SZ - 1) {
        float o = 0.f;
#pragma unroll
        for (int i = 0; i < DK_SZ; ++i) o = fmaf(sq[tt][i], S[i], o);
        olast[(size_t)bh * DV_SZ + col] = o;
      }
    }
  }
}

// ---------------------------------------------------------------------------
// k4: last-token gated RMSNorm: og[b][h*128+dv]
// ---------------------------------------------------------------------------
__global__ __launch_bounds__(128)
void k4_gate_norm(const float* __restrict__ olast, const float* __restrict__ z,
                  const float* __restrict__ Wg, const float* __restrict__ norm_w,
                  float* __restrict__ og) {
  __shared__ float red[128];
  const int bh = blockIdx.x;
  const int b = bh / H_SZ, h = bh - b * H_SZ;
  const int tid = threadIdx.x;
  const float o = olast[(size_t)bh * DV_SZ + tid];
  red[tid] = o * o;
  __syncthreads();
  for (int s = 64; s > 0; s >>= 1) {
    if (tid < s) red[tid] += red[tid + s];
    __syncthreads();
  }
  const float rn = rsqrtf(red[0] / 128.f + 1e-5f);
  const float* zl = z + (size_t)(b * T_SZ + T_SZ - 1) * D_SZ;
  const int col = h * DV_SZ + tid;
  float gacc = 0.f;
  for (int d = 0; d < D_SZ; ++d)
    gacc = fmaf(zl[d], Wg[(size_t)d * VAL_DIM + col], gacc);
  float val = o * rn * norm_w[tid];
  val = val * gacc * (1.f / (1.f + __expf(-gacc)));
  og[(size_t)b * VAL_DIM + col] = val;
}

// ---------------------------------------------------------------------------
// k5: output projection y[b][d] = og[b][:] @ Wo[:, d]
// ---------------------------------------------------------------------------
__global__ __launch_bounds__(256)
void k5_outproj(const float* __restrict__ og, const float* __restrict__ Wo,
                float* __restrict__ out) {
  const int b = blockIdx.y;
  const int d = blockIdx.x * 256 + threadIdx.x;
  const float* row = og + (size_t)b * VAL_DIM;
  float acc = 0.f;
  for (int c = 0; c < VAL_DIM; ++c)
    acc = fmaf(row[c], Wo[(size_t)c * D_SZ + d], acc);
  out[(size_t)b * D_SZ + d] = acc;
}

// ---------------------------------------------------------------------------
extern "C" void kernel_launch(void* const* d_in, const int* in_sizes, int n_in,
                              void* d_out, int out_size, void* d_ws, size_t ws_size,
                              hipStream_t stream) {
  const float* z       = (const float*)d_in[0];
  const float* Wq      = (const float*)d_in[1];
  const float* Wk      = (const float*)d_in[2];
  const float* Wv      = (const float*)d_in[3];
  const float* cq      = (const float*)d_in[4];
  const float* ck      = (const float*)d_in[5];
  const float* cv      = (const float*)d_in[6];
  const float* Wb      = (const float*)d_in[7];
  const float* Wa      = (const float*)d_in[8];
  const float* A_log   = (const float*)d_in[9];
  const float* dt_bias = (const float*)d_in[10];
  const float* Wg      = (const float*)d_in[11];
  const float* norm_w  = (const float*)d_in[12];
  const float* Wo      = (const float*)d_in[13];
  float* out = (float*)d_out;

  char* ws = (char*)d_ws;
  size_t off = 0;
  auto take = [&](size_t bytes) { char* p = ws + off; off += (bytes + 255) & ~(size_t)255; return p; };
  half_t* zf  = (half_t*)take((size_t)NTOK * D_SZ * sizeof(half_t));
  half_t* wf  = (half_t*)take((size_t)D_SZ * NPAD * sizeof(half_t));
  half_t* pre = (half_t*)take((size_t)NTOK * NPAD * sizeof(half_t));
  half_t* qb  = (half_t*)take((size_t)B_SZ * H_SZ * T_SZ * DK_SZ * sizeof(half_t));
  half_t* kb  = (half_t*)take((size_t)B_SZ * H_SZ * T_SZ * DK_SZ * sizeof(half_t));
  half_t* vb  = (half_t*)take((size_t)B_SZ * H_SZ * T_SZ * DV_SZ * sizeof(half_t));
  float*  gb  = (float*)take((size_t)B_SZ * H_SZ * T_SZ * sizeof(float));
  float*  bb  = (float*)take((size_t)B_SZ * H_SZ * T_SZ * sizeof(float));
  float*  ol  = (float*)take((size_t)B_SZ * H_SZ * DV_SZ * sizeof(float));
  float*  ogb = (float*)take((size_t)B_SZ * VAL_DIM * sizeof(float));

  k0_convert<<<4096, 256, 0, stream>>>(z, Wq, Wk, Wv, Wb, Wa, zf, wf);
  k1_gemm_fused<<<dim3(NPAD / 128, NTOK / 128), 256, 0, stream>>>(zf, wf, pre);
  k2_pointwise<<<NTOK, 256, 0, stream>>>(pre, cq, ck, cv, A_log, dt_bias,
                                         qb, kb, vb, gb, bb);
  k3_scan<<<B_SZ * H_SZ * NSPLIT, 32, 0, stream>>>(qb, kb, vb, gb, bb, ol);
  k4_gate_norm<<<B_SZ * H_SZ, 128, 0, stream>>>(ol, z, Wg, norm_w, ogb);
  k5_outproj<<<dim3(D_SZ / 256, B_SZ), 256, 0, stream>>>(ogb, Wo, out);
}
